// Attention_9234179687166
// MI455X (gfx1250) — matrix operand order repaired and checked by reading, not compiled, not run on hardware
//
#include <hip/hip_runtime.h>

// Problem constants (from the reference): B=32, L=K=D=1024
#define BDIM  32
#define LDIM  1024
#define KCTX  1024
#define DDIM  1024
#define MROWS (BDIM * LDIM)   // 32768
#define TWOD  (2 * DDIM)      // 2048

typedef __bf16 bf16;
typedef __attribute__((ext_vector_type(16))) __bf16 v16bf;
typedef __attribute__((ext_vector_type(8)))  __bf16 v8bf;
typedef __attribute__((ext_vector_type(2)))  __bf16 v2bf;
typedef __attribute__((ext_vector_type(8)))  float  v8f;
typedef __attribute__((ext_vector_type(4)))  unsigned uint4v;
typedef __attribute__((ext_vector_type(8)))  int    int8v;
typedef __attribute__((ext_vector_type(4)))  int    int4v;

// ---- CDNA5 feature probes (graceful fallback if a builtin is absent) ----
#if defined(__has_builtin)
#if __has_builtin(__builtin_amdgcn_global_load_async_to_lds_b128) && \
    __has_builtin(__builtin_amdgcn_s_wait_asynccnt)
#define HAVE_ASYNC_LDS 1
#endif
#if __has_builtin(__builtin_amdgcn_tensor_load_to_lds) && \
    __has_builtin(__builtin_amdgcn_s_wait_tensorcnt)
#define HAVE_TDM 1
#endif
#if __has_builtin(__builtin_amdgcn_cvt_pk_bf16_f32)
#define HAVE_PK_CVT 1
#endif
#endif

// async-to-LDS builtin takes (global int4*, local int4*, imm offset, imm cpol)
typedef __attribute__((address_space(1))) int4v gv4i;
typedef __attribute__((address_space(3))) int4v lv4i;
#define TO_GLOBAL(p) ((gv4i*)(unsigned long long)(p))
// Generic LDS pointers encode the LDS byte offset in the low 32 bits.
#define TO_LDS(p)    ((lv4i*)(unsigned)(unsigned long long)(p))

// Native fptrunc (RNE) — lets the backend pick the hardware bf16 convert.
__device__ __forceinline__ bf16 to_bf16(float f) { return (bf16)f; }

__device__ __forceinline__ v2bf pk2(float x, float y) {
#if defined(HAVE_PK_CVT)
    return __builtin_bit_cast(v2bf, __builtin_amdgcn_cvt_pk_bf16_f32(x, y));
#else
    v2bf r; r[0] = to_bf16(x); r[1] = to_bf16(y); return r;
#endif
}

union bfpack { v16bf v; v2bf p[8]; };

__device__ __forceinline__ v16bf cvt16(const float4* src) {
    bfpack u;
#pragma unroll
    for (int q = 0; q < 4; ++q) {
        float4 f = src[q];
        u.p[q * 2 + 0] = pk2(f.x, f.y);
        u.p[q * 2 + 1] = pk2(f.z, f.w);
    }
    return u.v;
}

__device__ __forceinline__ v16bf cat8(v8bf lo, v8bf hi) {
    return __builtin_shufflevector(lo, hi, 0,1,2,3,4,5,6,7,8,9,10,11,12,13,14,15);
}

__device__ __forceinline__ v8f wmma_bf16(v16bf a, v16bf b, v8f c) {
    return __builtin_amdgcn_wmma_f32_16x16x32_bf16(false, a, false, b, (short)0, c,
                                                   false, false);
}

// ---------------------------------------------------------------------------
// Tiled bf16 WMMA GEMM:  C[M,N=1024] = act( A[M,KDIM] * W[N,KDIM]^T )
// Block tile 128x256, 8 waves, each wave 64x64 (4x4 C fragments).
// W is f32 in HBM; converted to bf16 during LDS staging.
// FINAL=false: A is f32 (query), output bf16 into combined[:, col_offset+..]
// FINAL=true : A is bf16 (combined): staged via GLOBAL_LOAD_ASYNC_TO_LDS;
//              output f32 = tanh(acc)
// ---------------------------------------------------------------------------
template <typename AT, bool FINAL>
__global__ void __launch_bounds__(256)
gemm_bf16_wmma(const AT* __restrict__ A, int lda,
               const float* __restrict__ W, int ldw, int KDIM,
               void* __restrict__ Cout, int ldc, int col_offset)
{
    __shared__ bf16 Asub[128 * 32];
    __shared__ bf16 Bsub[256 * 32];

    const int tid  = threadIdx.x;
    const int lane = tid & 31;
    const int w    = tid >> 5;       // 0..7
    const int wm   = w & 1;          // row group of wave (0/1)
    const int wn   = w >> 1;         // col group of wave (0..3)
    const int m0   = blockIdx.y * 128;
    const int n0   = blockIdx.x * 256;

    v8f acc[4][4];
#pragma unroll
    for (int i = 0; i < 4; ++i)
#pragma unroll
        for (int j = 0; j < 4; ++j)
            acc[i][j] = v8f{0,0,0,0,0,0,0,0};

    const int aoff  = (lane & 16) ? 16 : 0;   // A frag k-offset (contiguous 16)
    const int boff  = (lane & 16) ? 16 : 0;   // B frag k-offset
    const int lm    = lane & 15;

    for (int k0 = 0; k0 < KDIM; k0 += 32) {
        // prefetch next K-step tiles into GL2 while this step computes
        if (k0 + 32 < KDIM) {
            __builtin_prefetch(&W[(size_t)(n0 + tid) * ldw + k0 + 32], 0, 1);
            __builtin_prefetch(&A[(size_t)(m0 + (tid >> 1)) * lda + k0 + 32], 0, 1);
        }
        __syncthreads();
        // ---- stage A tile (128x32) ----
        {
            const int r = tid >> 1, h = (tid & 1) * 16;
            const AT* gsrc = &A[(size_t)(m0 + r) * lda + k0 + h];
            bf16* ldst = &Asub[r * 32 + h];
            if constexpr (FINAL) {
#if defined(HAVE_ASYNC_LDS)
                // bf16 source: 32B async DMA to LDS (ASYNCcnt-tracked)
                __builtin_amdgcn_global_load_async_to_lds_b128(
                    TO_GLOBAL(gsrc), TO_LDS(ldst), 0, 0);
                __builtin_amdgcn_global_load_async_to_lds_b128(
                    TO_GLOBAL((const char*)gsrc + 16), TO_LDS((char*)ldst + 16), 0, 0);
#else
                *(v16bf*)ldst = *(const v16bf*)gsrc;
#endif
            } else {
                *(v16bf*)ldst = cvt16((const float4*)gsrc);
            }
        }
        // ---- stage B tile: W rows n0..n0+255, cols k0..k0+31 (f32 -> bf16) ----
        {
            const float4* src = (const float4*)&W[(size_t)(n0 + tid) * ldw + k0];
            *(v16bf*)&Bsub[tid * 32 + 0]  = cvt16(src);
            *(v16bf*)&Bsub[tid * 32 + 16] = cvt16(src + 4);
        }
#if defined(HAVE_ASYNC_LDS)
        if constexpr (FINAL) __builtin_amdgcn_s_wait_asynccnt(0);
#endif
        __syncthreads();

        // ---- fragments + 16 WMMAs ----
        v16bf av[4], bv[4];
#pragma unroll
        for (int mt = 0; mt < 4; ++mt) {
            const bf16* ap = &Asub[(wm * 64 + mt * 16 + lm) * 32];
            av[mt] = cat8(*(const v8bf*)(ap + aoff), *(const v8bf*)(ap + aoff + 8));
        }
#pragma unroll
        for (int nt = 0; nt < 4; ++nt)
            bv[nt] = *(const v16bf*)&Bsub[(wn * 64 + nt * 16 + lm) * 32 + boff];
#pragma unroll
        for (int mt = 0; mt < 4; ++mt)
#pragma unroll
            for (int nt = 0; nt < 4; ++nt)
                acc[mt][nt] = wmma_bf16(av[mt], bv[nt], acc[mt][nt]);
    }

    // ---- epilogue ----
    const int colb = n0 + wn * 64 + lm;
    const int rowb = m0 + wm * 64 + ((lane & 16) ? 8 : 0);
#pragma unroll
    for (int mt = 0; mt < 4; ++mt)
#pragma unroll
        for (int nt = 0; nt < 4; ++nt)
#pragma unroll
            for (int j = 0; j < 8; ++j) {
                const int row = rowb + mt * 16 + j;
                const int col = colb + nt * 16;
                float v = acc[mt][nt][j];
                if constexpr (FINAL)
                    ((float*)Cout)[(size_t)row * ldc + col] = tanhf(v);
                else
                    ((bf16*)Cout)[(size_t)row * ldc + col + col_offset] = to_bf16(v);
            }
}

// ---------------------------------------------------------------------------
// Flash-style masked attention: one block per 16 query rows.
// Online softmax of (scores * mask) with renormalized multiplicative mask,
// exactly matching: w = softmax(s*m)*m ; w /= (sum(w)+1e-13).
// LDS (~300 KB of the 320 KB/WGP): q tile (bf16, loaded by the Tensor Data
// Mover), transposed ctx chunk (bf16), S scratch (f32), P (bf16), stats.
// Wave w owns output columns d in [w*128, w*128+128) -> 8 C fragments.
// ---------------------------------------------------------------------------
__global__ void __launch_bounds__(256)
flash_attn(const float* __restrict__ ctx, const int* __restrict__ mask,
           bf16* __restrict__ combined)
{
    extern __shared__ __align__(16) char smem[];
    bf16*  qs   = (bf16*)smem;                   // 16 x 1024
    bf16*  ctxT = qs + 16 * 1024;                // 1024 x 128 (d-major, kk minor)
    float* Ssc  = (float*)(ctxT + 1024 * 128);   // 16 x 128
    bf16*  Pp   = (bf16*)(Ssc + 16 * 128);       // 16 x 128
    float* mrun = (float*)(Pp + 16 * 128);       // 16
    float* lrun = mrun + 16;                     // 16
    float* alph = lrun + 16;                     // 16

    const int tid  = threadIdx.x;
    const int lane = tid & 31;
    const int w    = tid >> 5;                   // 0..7
    const int lm   = lane & 15;
    const int rb   = (lane & 16) ? 8 : 0;        // C-frag row base
    const int aoff = (lane & 16) ? 16 : 0;
    const int boff = (lane & 16) ? 16 : 0;

    const int row0 = blockIdx.x * 16;            // query rows [row0, row0+16)
    const int bat  = row0 >> 10;                 // / L (=1024)

#if defined(HAVE_TDM)
    // ---- q tile via Tensor Data Mover: 2D tile 16 x 1024 bf16, row stride 2048
    // D# packing per CDNA5 ISA.
    if (w == 0) {
        const unsigned long long ga =
            (unsigned long long)(combined + (size_t)row0 * TWOD + DDIM);
        const unsigned lds_off = (unsigned)(unsigned long long)(void*)qs;
        uint4v g0;
        g0[0] = 1u;                                     // count=1, user descriptor
        g0[1] = lds_off;                                // lds_addr (bytes)
        g0[2] = (unsigned)(ga & 0xFFFFFFFFull);         // global_addr[31:0]
        g0[3] = (unsigned)(ga >> 32) | (2u << 30);      // global_addr[56:32] | type=2
        int8v g1;
        g1[0] = (int)(1u << 16);        // wg_mask=0, data_size=1 (2 bytes)
        g1[1] = (int)(1024u << 16);     // tensor_dim0[15:0]=1024 at bits 63:48
        g1[2] = (int)(16u << 16);       // tensor_dim1[15:0]=16 at bits 95:80
        g1[3] = (int)(1024u << 16);     // tile_dim0=1024 at bits 127:112
        g1[4] = 16;                     // tile_dim1=16, tile_dim2=0
        g1[5] = 2048;                   // tensor_dim0_stride[31:0]=2048
        g1[6] = 0;                      // stride hi / tensor_dim1_stride lo
        g1[7] = 0;
        int4v z4 = {0, 0, 0, 0};
#if (__clang_major__ >= 23)
        int8v z8 = {0, 0, 0, 0, 0, 0, 0, 0};
        __builtin_amdgcn_tensor_load_to_lds(g0, g1, z4, z4, z8, 0);
#else
        __builtin_amdgcn_tensor_load_to_lds(g0, g1, z4, z4, 0);
#endif
        __builtin_amdgcn_s_wait_tensorcnt(0);
    }
#else
    for (int i = tid; i < 16 * 64; i += 256) {
        const int r = i >> 6, c = (i & 63) * 16;
        *(v16bf*)&qs[r * 1024 + c] =
            *(const v16bf*)&combined[(size_t)(row0 + r) * TWOD + DDIM + c];
    }
#endif
    if (tid < 16) { mrun[tid] = -1e30f; lrun[tid] = 0.0f; }

    v8f acc[8];
#pragma unroll
    for (int t = 0; t < 8; ++t) acc[t] = v8f{0,0,0,0,0,0,0,0};

    const bf16* qrow = qs + lm * 1024;

    for (int kk0 = 0; kk0 < KCTX; kk0 += 128) {
        // prefetch next chunk of context into GL2
        if (kk0 + 128 < KCTX)
            __builtin_prefetch(
                &ctx[(size_t)(bat * KCTX + kk0 + 128 + (tid >> 1)) * DDIM + (tid & 1) * 512],
                0, 1);
        __syncthreads();   // prior chunk's ctxT / Pp consumers done
        // ---- stage transposed ctx chunk (f32 global -> bf16 LDS, d-major) ----
        for (int i = tid; i < 128 * 256; i += 256) {        // 128 rows x 256 float4
            const int kk = i >> 8, dq = (i & 255) << 2;
            float4 f = *(const float4*)&ctx[((size_t)(bat * KCTX + kk0 + kk)) * DDIM + dq];
            ctxT[(dq + 0) * 128 + kk] = to_bf16(f.x);
            ctxT[(dq + 1) * 128 + kk] = to_bf16(f.y);
            ctxT[(dq + 2) * 128 + kk] = to_bf16(f.z);
            ctxT[(dq + 3) * 128 + kk] = to_bf16(f.w);
        }
        __syncthreads();

        // ---- scores: wave w computes S tile, chunk columns [w*16, w*16+16) ----
        v8f s = v8f{0,0,0,0,0,0,0,0};
        const float* crow = ctx + (size_t)(bat * KCTX + kk0 + w * 16 + lm) * DDIM;
        for (int d0 = 0; d0 < DDIM; d0 += 32) {
            v16bf a = cat8(*(const v8bf*)(qrow + d0 + aoff),
                           *(const v8bf*)(qrow + d0 + aoff + 8));
            v16bf bb = cvt16((const float4*)(crow + d0 + boff));
            s = wmma_bf16(a, bb, s);
        }
        // apply multiplicative mask; masked -> -1e30 (exp -> 0), max clamps at 0
        const int colg = kk0 + w * 16 + lm;
#pragma unroll
        for (int j = 0; j < 8; ++j) {
            const int mrow = row0 + rb + j;
            const int mk = mask[(size_t)mrow * KCTX + colg];
            Ssc[(rb + j) * 128 + w * 16 + lm] = mk ? s[j] : -1e30f;
        }
        __syncthreads();

        // ---- online-softmax row stats (16 rows, one lane each) ----
        if (tid < 16) {
            const float* sr = &Ssc[tid * 128];
            float cm = 0.0f;                      // masked entries count as 0
            for (int c = 0; c < 128; ++c) cm = fmaxf(cm, sr[c]);
            const float mo = mrun[tid];
            const float mn = fmaxf(mo, cm);
            const float al = __expf(mo - mn);
            float ls = 0.0f;
            bf16* pr = &Pp[tid * 128];
            for (int c = 0; c < 128; ++c) {
                float p = __expf(sr[c] - mn);
                ls += p;
                pr[c] = to_bf16(p);
            }
            mrun[tid] = mn;
            lrun[tid] = lrun[tid] * al + ls;
            alph[tid] = al;
        }
        __syncthreads();

        // ---- rescale accumulators, then O += P * ctx_chunk ----
        float av[8];
#pragma unroll
        for (int j = 0; j < 8; ++j) av[j] = alph[rb + j];
#pragma unroll
        for (int t = 0; t < 8; ++t)
#pragma unroll
            for (int j = 0; j < 8; ++j) acc[t][j] *= av[j];

        const bf16* prow = Pp + lm * 128;
        for (int kc = 0; kc < 128; kc += 32) {
            v16bf pa = cat8(*(const v8bf*)(prow + kc + aoff),
                            *(const v8bf*)(prow + kc + aoff + 8));
#pragma unroll
            for (int nt = 0; nt < 8; ++nt) {
                const int drow = w * 128 + nt * 16 + lm;
                v16bf bb = *(const v16bf*)&ctxT[drow * 128 + kc + boff];
                acc[nt] = wmma_bf16(pa, bb, acc[nt]);
            }
        }
    }
    __syncthreads();

    // ---- normalize and store mix (bf16) into combined[:, 0:D] ----
    float rv[8];
#pragma unroll
    for (int j = 0; j < 8; ++j) rv[j] = 1.0f / (lrun[rb + j] + 1e-13f);
#pragma unroll
    for (int nt = 0; nt < 8; ++nt)
#pragma unroll
        for (int j = 0; j < 8; ++j) {
            const int row = row0 + rb + j;
            const int col = w * 128 + nt * 16 + lm;
            combined[(size_t)row * TWOD + col] = to_bf16(acc[nt][j] * rv[j]);
        }
}

// ---------------------------------------------------------------------------
extern "C" void kernel_launch(void* const* d_in, const int* in_sizes, int n_in,
                              void* d_out, int out_size, void* d_ws, size_t ws_size,
                              hipStream_t stream)
{
    (void)in_sizes; (void)n_in; (void)out_size; (void)ws_size;
    const float* query   = (const float*)d_in[0];   // [B,L,D] f32
    const float* context = (const float*)d_in[1];   // [B,K,D] f32
    const int*   mask    = (const int*)d_in[2];     // [B*L,K] i32
    const float* W_in    = (const float*)d_in[3];   // [D,D]   f32
    const float* W_out   = (const float*)d_in[4];   // [D,2D]  f32
    float* out = (float*)d_out;                     // [B,L,D] f32

    bf16* combined = (bf16*)d_ws;                   // [M, 2D] bf16 = 128 MB

    dim3 gg(DDIM / 256, MROWS / 128);               // (4, 256)

    // 1) q = query @ W_in^T  -> combined[:, D:2D] (bf16)
    gemm_bf16_wmma<float, false><<<gg, 256, 0, stream>>>(
        query, DDIM, W_in, DDIM, DDIM, combined, TWOD, DDIM);

    // 2) flash masked attention -> combined[:, 0:D] (bf16)
    const size_t lds_bytes = (size_t)16 * 1024 * 2   // qs
                           + (size_t)1024 * 128 * 2  // ctxT
                           + (size_t)16 * 128 * 4    // Ssc
                           + (size_t)16 * 128 * 2    // Pp
                           + 3 * 16 * 4;             // stats
    flash_attn<<<MROWS / 16, 256, lds_bytes, stream>>>(context, mask, combined);

    // 3) out = tanh(combined @ W_out^T) (f32)
    gemm_bf16_wmma<bf16, true><<<gg, 256, 0, stream>>>(
        combined, TWOD, W_out, TWOD, TWOD, out, DDIM, 0);
}
